// RNN_7086696038959
// MI455X (gfx1250) — matrix operand order repaired and checked by reading, not compiled, not run on hardware
//
#include <hip/hip_runtime.h>

// ---------------------------------------------------------------------------
// Sizes (match reference)
#define VOCAB   80
#define EMB     100
#define HID     100
#define LAYERS  8
#define BSZ     512
#define SEQ     1024
#define HIDP    112          // HID padded to 7*16
#define KPAD    224          // (EMB->112) + (HID->112)
#define NTILES  7            // 112/16
#define KTILES  7            // 224/32

typedef __attribute__((ext_vector_type(16))) __bf16 v16bf;
typedef __attribute__((ext_vector_type(8)))  float  v8f;

// Branch-free tanh on the native v_exp_f32 / v_rcp_f32 path.
// tanh(x) = sign(x) * (1 - 2 / (exp(2|x|) + 1)); |x| large -> exp=inf -> rcp=0 -> 1.
__device__ __forceinline__ float fast_tanh(float x) {
    float ax = __builtin_fabsf(x);
    float e  = __expf(ax * 2.0f);                       // v_mul + v_exp_f32
    float r  = __builtin_amdgcn_rcpf(e + 1.0f);         // v_rcp_f32
    float t  = __builtin_fmaf(-2.0f, r, 1.0f);
    return __builtin_copysignf(t, x);
}

// ---------------------------------------------------------------------------
// Pack W_ih / W_hh into WMMA B-fragment order (bf16), and fuse biases.
// B-fragment, this code's contiguous k order (same as A): lane L: N = L%16 ; elem e: K = (L/16)*16 + e
__global__ void rnn_pack_kernel(const float* __restrict__ W_ih,
                                const float* __restrict__ W_hh,
                                const float* __restrict__ b_ih,
                                const float* __restrict__ b_hh,
                                __bf16* __restrict__ packW,
                                float*  __restrict__ packBias) {
    int i = blockIdx.x * blockDim.x + threadIdx.x;
    const int total = LAYERS * NTILES * KTILES * 32 * 16;
    if (i < total) {
        int e    =  i        & 15;
        int lane = (i >> 4)  & 31;
        int rem  =  i >> 9;             // (l*7+n)*7 + k
        int k    =  rem % KTILES;
        int n    = (rem / KTILES) % NTILES;
        int l    =  rem / (KTILES * NTILES);
        int K    = k * 32 + ((lane >> 4) << 4) + e;
        int N    = n * 16 + (lane & 15);
        float v = 0.0f;
        if (N < HID) {
            if (K < EMB)                      v = W_ih[((size_t)l * HID + N) * EMB + K];
            else if (K >= 112 && K < 112+HID) v = W_hh[((size_t)l * HID + N) * HID + (K - 112)];
        }
        packW[i] = (__bf16)v;
    }
    if (i < LAYERS * HIDP) {
        int l = i / HIDP, c = i % HIDP;
        packBias[i] = (c < HID) ? (b_ih[l * HID + c] + b_hh[l * HID + c]) : 0.0f;
    }
}

// ---------------------------------------------------------------------------
// Embedding gather + LayerNorm -> bf16 sequence buffer, layout seq[t][b][112]
__global__ void emb_ln_kernel(const int* __restrict__ x,
                              const float* __restrict__ emb,
                              const float* __restrict__ gamma,
                              const float* __restrict__ beta,
                              __bf16* __restrict__ seq) {
    size_t i = (size_t)blockIdx.x * blockDim.x + threadIdx.x;
    if (i >= (size_t)BSZ * SEQ) return;
    int t = (int)(i / BSZ);
    int b = (int)(i % BSZ);
    int idx = x[(size_t)b * SEQ + t];
    const float* e = emb + (size_t)idx * EMB;
    float mu = 0.0f;
    for (int k = 0; k < EMB; ++k) mu += e[k];
    mu *= (1.0f / EMB);
    float var = 0.0f;
    for (int k = 0; k < EMB; ++k) { float d = e[k] - mu; var += d * d; }
    var *= (1.0f / EMB);
    float rs = rsqrtf(var + 1e-5f);
    __bf16* o = seq + i * HIDP;
    for (int k = 0; k < EMB; ++k) o[k] = (__bf16)((e[k] - mu) * rs * gamma[k] + beta[k]);
    for (int k = EMB; k < HIDP; ++k) o[k] = (__bf16)0.0f;
}

// ---------------------------------------------------------------------------
// Persistent layer-major RNN. One WG per 16 batch rows; 7 waves = 7 N-tiles.
// Software-pipelined; A-fragments in distinct regs (overlapped DS waits);
// dual WMMA accumulator chains; 2 barriers per step.
__global__ __launch_bounds__(224) void rnn_core_kernel(
        const __bf16* __restrict__ packW,
        const float*  __restrict__ packBias,
        __bf16* __restrict__ seq,           // in-place [t][b][112] bf16
        float*  __restrict__ out_hidden) {  // (LAYERS, B, HID) f32
    __shared__ __bf16 A[16][KPAD];          // [inp(112) | h(112)] per batch row

    const int tid  = threadIdx.x;
    const int wave = tid >> 5;              // = N-tile index, 0..6
    const int lane = tid & 31;
    const int b0   = blockIdx.x * 16;

    // A-fragment geometry (16-bit A 16x32 layout)
    const int aM  = lane & 15;
    const int aK0 = (lane >> 4) << 4;       // 0 or 16
    // C/D geometry
    const int nCol    = lane & 15;
    const int mBase   = (lane >> 4) << 3;   // 0 or 8
    const int colBase = wave * 16 + nCol;   // 0..111

    // loader geometry: 224 threads x 16B = one 16x112 bf16 tile
    const int ldRow = tid / 14;
    const int ldChk = tid % 14;

    for (int l = 0; l < LAYERS; ++l) {
        // ---- preload this layer's B-fragments into registers (stay for all t)
        v16bf Bfrag[KTILES];
#pragma unroll
        for (int k = 0; k < KTILES; ++k) {
            const __bf16* p = packW +
                ((((size_t)l * NTILES + wave) * KTILES + k) * 32 + lane) * 16;
            Bfrag[k] = *(const v16bf*)p;
        }
        const float bias = packBias[l * HIDP + colBase];

        float hprev[8] = {0.f, 0.f, 0.f, 0.f, 0.f, 0.f, 0.f, 0.f};  // h(-1)
        // prefetch inp(t=0)
        uint4 vin = *(const uint4*)(seq + ((size_t)0 * BSZ + b0 + ldRow) * HIDP + ldChk * 8);

        for (int t = 0; t < SEQ; ++t) {
            // ---- write phase: stage inp(t) + h(t-1) into LDS
            *(uint4*)&A[ldRow][ldChk * 8] = vin;
#pragma unroll
            for (int r = 0; r < 8; ++r)
                A[mBase + r][HIDP + colBase] = (__bf16)hprev[r];
            // prefetch inp(t+1) now: hidden under barrier + whole compute phase
            if (t + 1 < SEQ)
                vin = *(const uint4*)(seq + ((size_t)(t + 1) * BSZ + b0 + ldRow) * HIDP + ldChk * 8);
            __syncthreads();

            // ---- read/compute phase
            if (t > 0) {         // coalesced flush: h(t-1) (in LDS) -> seq[t-1]
                const uint4 hv = *(const uint4*)&A[ldRow][HIDP + ldChk * 8];
                *(uint4*)(seq + ((size_t)(t - 1) * BSZ + b0 + ldRow) * HIDP + ldChk * 8) = hv;
            }

            // load all A-fragments into distinct registers (pipelined DS waits)
            v16bf afrag[KTILES];
#pragma unroll
            for (int k = 0; k < KTILES; ++k) {
                const __bf16* pa = &A[aM][k * 32 + aK0];
                ((uint4*)&afrag[k])[0] = *(const uint4*)pa;          // K0..K0+7
                ((uint4*)&afrag[k])[1] = *(const uint4*)(pa + 8);    // K0+8..K0+15
            }

            // two independent accumulator chains (halved WMMA RAW depth)
            v8f acc0 = { bias, bias, bias, bias, bias, bias, bias, bias };
            v8f acc1 = { 0.f, 0.f, 0.f, 0.f, 0.f, 0.f, 0.f, 0.f };
#pragma unroll
            for (int k = 0; k < 4; ++k)
                acc0 = __builtin_amdgcn_wmma_f32_16x16x32_bf16(
                        false, afrag[k], false, Bfrag[k], (short)0, acc0, false, false);
#pragma unroll
            for (int k = 4; k < KTILES; ++k)
                acc1 = __builtin_amdgcn_wmma_f32_16x16x32_bf16(
                        false, afrag[k], false, Bfrag[k], (short)0, acc1, false, false);

#pragma unroll
            for (int r = 0; r < 8; ++r) hprev[r] = fast_tanh(acc0[r] + acc1[r]);
            __syncthreads();
        }

        // ---- layer epilogue: h(SEQ-1) -> seq[SEQ-1] (+ f32 hidden-state output)
#pragma unroll
        for (int r = 0; r < 8; ++r) {
            const int M = r + mBase;
            seq[((size_t)(SEQ - 1) * BSZ + b0 + M) * HIDP + colBase] = (__bf16)hprev[r];
        }
        if (colBase < HID) {
#pragma unroll
            for (int r = 0; r < 8; ++r) {
                const int M = r + mBase;
                out_hidden[((size_t)l * BSZ + b0 + M) * HID + colBase] = hprev[r];
            }
        }
        __threadfence();       // this layer's seq stores visible to next layer's loads
        __syncthreads();
    }
}

// ---------------------------------------------------------------------------
// logits = out_last @ W_fc^T + b_fc  (512x100x80, trivially small)
__global__ void fc_kernel(const __bf16* __restrict__ seq,
                          const float* __restrict__ W_fc,
                          const float* __restrict__ b_fc,
                          float* __restrict__ logits) {
    int i = blockIdx.x * blockDim.x + threadIdx.x;
    if (i >= BSZ * VOCAB) return;
    int b = i / VOCAB, v = i % VOCAB;
    const __bf16* h = seq + (((size_t)(SEQ - 1) * BSZ) + b) * HIDP;
    float s = b_fc[v];
    for (int k = 0; k < HID; ++k) s += (float)h[k] * W_fc[(size_t)v * HID + k];
    logits[i] = s;
}

// ---------------------------------------------------------------------------
extern "C" void kernel_launch(void* const* d_in, const int* in_sizes, int n_in,
                              void* d_out, int out_size, void* d_ws, size_t ws_size,
                              hipStream_t stream) {
    (void)in_sizes; (void)n_in; (void)out_size; (void)ws_size;
    const int*   x     = (const int*)  d_in[0];
    const float* emb   = (const float*)d_in[1];
    const float* gamma = (const float*)d_in[2];
    const float* beta  = (const float*)d_in[3];
    const float* W_ih  = (const float*)d_in[4];
    const float* W_hh  = (const float*)d_in[5];
    const float* b_ih  = (const float*)d_in[6];
    const float* b_hh  = (const float*)d_in[7];
    const float* W_fc  = (const float*)d_in[8];
    const float* b_fc  = (const float*)d_in[9];
    float* out = (float*)d_out;   // [512*80 logits][8*512*100 hidden]

    char* ws = (char*)d_ws;
    const size_t seqBytes   = (size_t)BSZ * SEQ * HIDP * 2;                   // 117,440,512
    const size_t packWBytes = (size_t)LAYERS * NTILES * KTILES * 32 * 16 * 2; // 401,408
    __bf16* seq      = (__bf16*)ws;
    __bf16* packW    = (__bf16*)(ws + seqBytes);
    float*  packBias = (float*) (ws + seqBytes + packWBytes);

    {
        const int total = LAYERS * NTILES * KTILES * 32 * 16;
        rnn_pack_kernel<<<(total + 255) / 256, 256, 0, stream>>>(
            W_ih, W_hh, b_ih, b_hh, packW, packBias);
    }
    emb_ln_kernel<<<(BSZ * SEQ + 255) / 256, 256, 0, stream>>>(
        x, emb, gamma, beta, seq);
    rnn_core_kernel<<<BSZ / 16, 224, 0, stream>>>(
        packW, packBias, seq, out + BSZ * VOCAB);
    fc_kernel<<<(BSZ * VOCAB + 255) / 256, 256, 0, stream>>>(
        seq, W_fc, b_fc, out);
}
